// GATrNet_55130200212216
// MI455X (gfx1250) — matrix operand order repaired and checked by reading, not compiled, not run on hardware
//
#include <hip/hip_runtime.h>
#include <hip/hip_bf16.h>

// ---------------------------------------------------------------------------
// GATr block on gfx1250: blade-fused GEMMs + flash-style attention, all matmul
// through v_wmma_f32_16x16x32_f16 (wave32 WMMA, f32 accumulation).
// Round 3: GEMM templated on K/mode so B-fragment addresses fold into the
// 24-bit load immediates (restores software-pipelined loads, kills per-iter
// v_add_nc_u64 + full loadcnt waits seen in round 2).
// ---------------------------------------------------------------------------

typedef __attribute__((ext_vector_type(16))) _Float16 v16h;
typedef __attribute__((ext_vector_type(8)))  _Float16 v8h;
typedef __attribute__((ext_vector_type(8)))  float    v8f;

#define BB_    4
#define SS_    2048
#define CIN_   3
#define MV_    16
#define HID_   32
#define NH_    8
#define NBLADE_ 9
#define MROWS_ (BB_*SS_)      // 8192 tokens
#define KQ_    256            // per-head q/k dim = HID*8 inner coords
#define DV_    512            // HID*MV (v dim, shared across heads)

__constant__ int IC_[8] = {0,2,3,4,8,9,10,14};   // non-degenerate blade coords

// --- WMMA helpers -----------------------------------------------------------
// A fragment (16x32 f16): lane<16 -> row=lane,   K = {0..15}
//                         lane>=16-> row=lane-16,K = {16..31}
// => per lane: 16 contiguous f16 at (k0 + 16*hs).
__device__ __forceinline__ v16h load_a16(const _Float16* p) {
  v8h lo = *(const v8h*)(p);
  v8h hi = *(const v8h*)(p + 8);
  v16h a;
#pragma unroll
  for (int i = 0; i < 8; ++i) { a[i] = lo[i]; a[i + 8] = hi[i]; }
  return a;
}
// B fragment (32x16 f16) with B stored transposed [N][K]:
// lane: col = lane%16, K-base = 16*(lane/16) -> 16 contiguous f16 = one v16h.
__device__ __forceinline__ v8f wmma_f16(v16h a, v16h b, v8f c) {
  return __builtin_amdgcn_wmma_f32_16x16x32_f16(false, a, false, b,
                                                (short)0, c, false, false);
}

// --- Kernel 1: fuse blade into a weight, write Wb^T [N][Kpad] f16 -----------
// WbT[n][k] = sum_b w[j, i, b] * blade[b, x, y]; k=(i,x); n->(j,y) per mode.
__global__ void fuse_w(const float* __restrict__ w, const float* __restrict__ blade,
                       _Float16* __restrict__ wbt, int N, int K, int Kpad,
                       int IN, int mode) {
  int idx = blockIdx.x * 256 + threadIdx.x;
  if (idx >= N * Kpad) return;
  int n = idx / Kpad, k = idx % Kpad;
  if (k >= K) { wbt[idx] = (_Float16)0.f; return; }
  int i = k >> 4, x = k & 15;
  int j, y;
  if (mode == 0)      { j = n >> 4; y = n & 15; }                       // plain
  else if (mode == 1) { int head = n >> 8, hid = (n >> 3) & 31, c = n & 7;
                        j = hid * NH_ + head; y = IC_[c]; }             // q-inner
  else                { int hid = n >> 3, c = n & 7;
                        j = hid; y = IC_[c]; }                          // k-inner
  float s = 0.f;
#pragma unroll
  for (int bb = 0; bb < NBLADE_; ++bb)
    s += w[(j * IN + i) * NBLADE_ + bb] * blade[bb * 256 + x * 16 + y];
  wbt[idx] = (_Float16)s;
}

// --- Kernel 2: x f32 [8192,48] -> f16 [8192,64] zero-padded -----------------
__global__ void conv_x(const float* __restrict__ x, _Float16* __restrict__ x16) {
  int idx = blockIdx.x * 256 + threadIdx.x;           // 8192*64
  int row = idx >> 6, k = idx & 63;
  x16[idx] = (k < CIN_ * MV_) ? (_Float16)x[row * (CIN_ * MV_) + k]
                              : (_Float16)0.f;
}

// --- Kernel 3: generic WMMA GEMM, wave computes 32x64 of C ------------------
// A [M][K] f16 row-major, Bt [N][K] f16.  K and epilogue mode are template
// parameters so every fragment address is base + compile-time immediate.
// MODE 0: f32 out; 1: f16 out; 2: f16 out = acc + Cadd (f32 residual add).
template <int K, int MODE>
__global__ void gemm_wmma(const _Float16* __restrict__ A,
                          const _Float16* __restrict__ Bt,
                          const float* __restrict__ Cadd,
                          float* __restrict__ out32,
                          _Float16* __restrict__ out16,
                          int M, int N) {
  const int lane = threadIdx.x & 31, wave = threadIdx.x >> 5;
  const int l16 = lane & 15, hs = lane >> 4;
  const int tn64 = N >> 6;
  const int id = blockIdx.x * 8 + wave;
  if (id >= (M >> 5) * tn64) return;
  const int tm = id / tn64, tn = id % tn64;

  // Single A base and single B base; all other offsets are constexpr.
  const _Float16* aB = A + (size_t)(tm * 32 + l16) * K + 16 * hs;
  const _Float16* bB = Bt + (size_t)(tn * 64 + l16) * K + 16 * hs;

  v8f acc[2][4] = {};
  for (int k0 = 0; k0 < K; k0 += 32) {
    v16h a0 = load_a16(aB + k0);
    v16h a1 = load_a16(aB + 16 * K + k0);
#pragma unroll
    for (int nt = 0; nt < 4; ++nt) {
      v16h b = *(const v16h*)(bB + nt * (16 * K) + k0);
      acc[0][nt] = wmma_f16(a0, b, acc[0][nt]);
      acc[1][nt] = wmma_f16(a1, b, acc[1][nt]);
    }
  }
#pragma unroll
  for (int ms = 0; ms < 2; ++ms) {
#pragma unroll
    for (int nt = 0; nt < 4; ++nt) {
      int col = tn * 64 + nt * 16 + l16;
#pragma unroll
      for (int r = 0; r < 8; ++r) {
        int row = tm * 32 + ms * 16 + r + 8 * hs;      // D layout: rows split by half-wave
        size_t o = (size_t)row * N + col;
        float v = acc[ms][nt][r];
        if (MODE == 0)      out32[o] = v;
        else if (MODE == 1) out16[o] = (_Float16)v;
        else                out16[o] = (_Float16)(v + Cadd[o]);
      }
    }
  }
}

// --- Kernel 4: equi_norm: per-token scale over inner coords -----------------
__global__ void equinorm_k(const float* __restrict__ h, _Float16* __restrict__ hn) {
  int lane = threadIdx.x & 31, wave = threadIdx.x >> 5;
  int row = blockIdx.x * 8 + wave;                     // one wave per token
  if (row >= MROWS_) return;
  const float* hr = h + (size_t)row * DV_ + lane * MV_;
  float ip = 0.f;
#pragma unroll
  for (int c = 0; c < 8; ++c) { float v = hr[IC_[c]]; ip += v * v; }
#pragma unroll
  for (int mk = 1; mk < 32; mk <<= 1) ip += __shfl_xor(ip, mk, 32);
  float scale = rsqrtf(ip * (1.0f / HID_));
  _Float16* hw = hn + (size_t)row * DV_ + lane * MV_;
#pragma unroll
  for (int t = 0; t < MV_; ++t) hw[t] = (_Float16)(hr[t] * scale);
}

// --- Kernel 5: V transpose per batch: vs[b][t][d] -> vt[b][d][t] ------------
__global__ void transp_v(const _Float16* __restrict__ vs, _Float16* __restrict__ vt) {
  int idx = blockIdx.x * 256 + threadIdx.x;            // 4 * 512 * 2048
  int t = idx & (SS_ - 1);
  int d = (idx >> 11) & (DV_ - 1);
  int b = idx >> 20;
  vt[idx] = vs[((size_t)(b * SS_ + t)) * DV_ + d];
}

// --- Kernel 6: attention, one block = (batch, head, 16-row q tile) ----------
// Pass A: compute S = scale*Q@K^T once, park it in LDS (128 KB of the WGP's
// 320 KB), track row max.  Pass B: read S, exponentiate, stage P (f16) for
// the PV WMMAs, accumulate the softmax denominator along the way, normalize
// at the end.  Scores never touch global memory; QK^T is computed once.
__global__ void attn_wmma(const _Float16* __restrict__ qs,   // [8192][2048]
                          const _Float16* __restrict__ ks,   // [8192][256]
                          const _Float16* __restrict__ vt,   // [4][512][2048]
                          _Float16* __restrict__ ao) {       // [8192][4096]
  __shared__ float s_lds[16 * SS_];                    // scaled scores, 128 KB
  __shared__ __align__(16) _Float16 p_lds[16 * 256];   // staged exp(P), 8 KB
  __shared__ float red_m[8][16], red_l[8][16];
  __shared__ float Mg[16], Lg[16];

  const int tid = threadIdx.x, lane = tid & 31, wave = tid >> 5;
  const int l16 = lane & 15, hs = lane >> 4;
  const int bid = blockIdx.x;
  const int qt = bid & 127, head = (bid >> 7) & 7, b = bid >> 10;
  const int q0 = qt * 16;
  const float scale = 0.0625f;                         // 1/sqrt(256)

  const _Float16* kbase = ks + (size_t)(b * SS_) * KQ_;

  // ---- Pass A: S tile + row max; wave w owns keys [w*256, w*256+256)
  {
    const _Float16* qrow =
        qs + (size_t)(b * SS_ + q0 + l16) * (NH_ * KQ_) + head * KQ_ + 16 * hs;
    v16h Qf[8];
#pragma unroll
    for (int kk = 0; kk < 8; ++kk) Qf[kk] = load_a16(qrow + kk * 32);

    float m[8];
#pragma unroll
    for (int r = 0; r < 8; ++r) m[r] = -1e30f;
    for (int ch = 0; ch < 8; ++ch) {
      int c0 = wave * 256 + ch * 32;
      v8f S0 = {}, S1 = {};
      const _Float16* kp0 = kbase + (size_t)(c0 + l16) * KQ_ + 16 * hs;
      const _Float16* kp1 = kbase + (size_t)(c0 + 16 + l16) * KQ_ + 16 * hs;
#pragma unroll
      for (int kk = 0; kk < 8; ++kk) {
        S0 = wmma_f16(Qf[kk], *(const v16h*)(kp0 + kk * 32), S0);
        S1 = wmma_f16(Qf[kk], *(const v16h*)(kp1 + kk * 32), S1);
      }
#pragma unroll
      for (int r = 0; r < 8; ++r) {
        float s0 = S0[r] * scale, s1 = S1[r] * scale;
        s_lds[(r + 8 * hs) * SS_ + c0 + l16]      = s0;
        s_lds[(r + 8 * hs) * SS_ + c0 + 16 + l16] = s1;
        float cm = fmaxf(s0, s1);
#pragma unroll
        for (int mk = 1; mk < 16; mk <<= 1) cm = fmaxf(cm, __shfl_xor(cm, mk, 32));
        m[r] = fmaxf(m[r], cm);
      }
    }
    if (l16 == 0) {
#pragma unroll
      for (int r = 0; r < 8; ++r) red_m[wave][r + 8 * hs] = m[r];
    }
  }
  __syncthreads();
  if (tid < 16) {
    float M = -1e30f;
    for (int w = 0; w < 8; ++w) M = fmaxf(M, red_m[w][tid]);
    Mg[tid] = M;
  }
  __syncthreads();
  float Mreg[8];
#pragma unroll
  for (int r = 0; r < 8; ++r) Mreg[r] = Mg[r + 8 * hs];

  // ---- Pass B: P = exp(S - M) from LDS, accumulate O = P @ V and row sums.
  v8f O[4] = {};
  float lsum[8];
#pragma unroll
  for (int r = 0; r < 8; ++r) lsum[r] = 0.f;
  const _Float16* vb = vt + (size_t)b * DV_ * SS_;
  for (int sc = 0; sc < 8; ++sc) {
    int c0 = sc * 256 + wave * 32;                     // this wave stages 32 keys
#pragma unroll
    for (int r = 0; r < 8; ++r) {
      float s0 = s_lds[(r + 8 * hs) * SS_ + c0 + l16];
      float s1 = s_lds[(r + 8 * hs) * SS_ + c0 + 16 + l16];
      float p0 = __expf(s0 - Mreg[r]);
      float p1 = __expf(s1 - Mreg[r]);
      lsum[r] += p0 + p1;
      p_lds[(r + 8 * hs) * 256 + wave * 32 + l16]      = (_Float16)p0;
      p_lds[(r + 8 * hs) * 256 + wave * 32 + 16 + l16] = (_Float16)p1;
    }
    __syncthreads();
    // PV: wave owns v-cols [wave*64, wave*64+64); P A-frags come from LDS.
    const _Float16* pl = p_lds + l16 * 256 + 16 * hs;
    for (int kk = 0; kk < 8; ++kk) {
      v16h pa = load_a16(pl + kk * 32);
#pragma unroll
      for (int nt = 0; nt < 4; ++nt) {
        int vc = wave * 64 + nt * 16 + l16;
        const _Float16* vp = vb + (size_t)vc * SS_ + sc * 256 + kk * 32 + 16 * hs;
        O[nt] = wmma_f16(pa, *(const v16h*)vp, O[nt]);
      }
    }
    __syncthreads();
  }
  // Combine per-wave partial row sums -> softmax denominator.
#pragma unroll
  for (int r = 0; r < 8; ++r) {
#pragma unroll
    for (int mk = 1; mk < 16; mk <<= 1) lsum[r] += __shfl_xor(lsum[r], mk, 32);
  }
  if (l16 == 0) {
#pragma unroll
    for (int r = 0; r < 8; ++r) red_l[wave][r + 8 * hs] = lsum[r];
  }
  __syncthreads();
  if (tid < 16) {
    float L = 0.f;
    for (int w = 0; w < 8; ++w) L += red_l[w][tid];
    Lg[tid] = L;
  }
  __syncthreads();
  float Lreg[8];
#pragma unroll
  for (int r = 0; r < 8; ++r) Lreg[r] = Lg[r + 8 * hs];

  // Normalize and scatter to [token][head*512 + d] for the out-projection GEMM.
#pragma unroll
  for (int nt = 0; nt < 4; ++nt) {
    int col = head * DV_ + wave * 64 + nt * 16 + l16;
#pragma unroll
    for (int r = 0; r < 8; ++r) {
      int row = b * SS_ + q0 + r + 8 * hs;
      ao[(size_t)row * (NH_ * DV_) + col] = (_Float16)(O[nt][r] / Lreg[r]);
    }
  }
}

// ---------------------------------------------------------------------------
extern "C" void kernel_launch(void* const* d_in, const int* in_sizes, int n_in,
                              void* d_out, int out_size, void* d_ws, size_t ws_size,
                              hipStream_t stream) {
  const float* x       = (const float*)d_in[0];
  const float* blade   = (const float*)d_in[1];
  const float* w_enter = (const float*)d_in[2];
  const float* w_q     = (const float*)d_in[3];
  const float* w_k     = (const float*)d_in[4];
  const float* w_v     = (const float*)d_in[5];
  const float* w_out   = (const float*)d_in[6];
  const float* w_final = (const float*)d_in[7];
  float* out = (float*)d_out;

  // Workspace carve (256B aligned).
  char* p = (char*)d_ws;
  auto take = [&](size_t bytes) { char* r = p; p += (bytes + 255) & ~(size_t)255; return r; };
  _Float16* wbt_enter = (_Float16*)take((size_t)512 * 64 * 2);
  _Float16* wbt_q     = (_Float16*)take((size_t)2048 * 512 * 2);
  _Float16* wbt_k     = (_Float16*)take((size_t)256 * 512 * 2);
  _Float16* wbt_v     = (_Float16*)take((size_t)512 * 512 * 2);
  _Float16* wbt_out   = (_Float16*)take((size_t)512 * 4096 * 2);
  _Float16* wbt_fin   = (_Float16*)take((size_t)512 * 512 * 2);
  _Float16* x16 = (_Float16*)take((size_t)MROWS_ * 64 * 2);
  float*    h   = (float*)   take((size_t)MROWS_ * DV_ * 4);
  _Float16* hn  = (_Float16*)take((size_t)MROWS_ * DV_ * 2);
  _Float16* qsb = (_Float16*)take((size_t)MROWS_ * 2048 * 2);
  _Float16* ksb = (_Float16*)take((size_t)MROWS_ * 256 * 2);
  _Float16* vsb = (_Float16*)take((size_t)MROWS_ * DV_ * 2);
  _Float16* vtb = (_Float16*)take((size_t)BB_ * DV_ * SS_ * 2);
  _Float16* ao  = (_Float16*)take((size_t)MROWS_ * 4096 * 2);
  _Float16* h2  = (_Float16*)take((size_t)MROWS_ * DV_ * 2);

  // 1) Fuse blade into all weights (Bt layout [N][Kpad] f16).
  auto fuse = [&](const float* w, _Float16* dst, int N, int K, int Kpad, int IN, int mode) {
    int n = N * Kpad;
    fuse_w<<<(n + 255) / 256, 256, 0, stream>>>(w, blade, dst, N, K, Kpad, IN, mode);
  };
  fuse(w_enter, wbt_enter, 512, 48, 64, CIN_, 0);
  fuse(w_q,     wbt_q,     2048, 512, 512, HID_, 1);
  fuse(w_k,     wbt_k,     256,  512, 512, HID_, 2);
  fuse(w_v,     wbt_v,     512,  512, 512, HID_, 0);
  fuse(w_out,   wbt_out,   512, 4096, 4096, HID_ * NH_, 0);
  fuse(w_final, wbt_fin,   512,  512, 512, HID_, 0);

  // 2) x -> f16, K padded 48 -> 64.
  conv_x<<<(MROWS_ * 64) / 256, 256, 0, stream>>>(x, x16);

  auto blocks = [](int M, int N) { return ((M / 32) * (N / 64) + 7) / 8; };

  // 3) h = x @ Wb_enter (f32, residual).
  gemm_wmma<64, 0><<<blocks(MROWS_, 512), 256, 0, stream>>>(
      x16, wbt_enter, nullptr, h, nullptr, MROWS_, 512);
  // 4) hn = equi_norm(h) (f16).
  equinorm_k<<<MROWS_ / 8, 256, 0, stream>>>(h, hn);
  // 5) q/k/v projections (inner-coord-pruned q/k).
  gemm_wmma<512, 1><<<blocks(MROWS_, 2048), 256, 0, stream>>>(
      hn, wbt_q, nullptr, nullptr, qsb, MROWS_, 2048);
  gemm_wmma<512, 1><<<blocks(MROWS_, 256), 256, 0, stream>>>(
      hn, wbt_k, nullptr, nullptr, ksb, MROWS_, 256);
  gemm_wmma<512, 1><<<blocks(MROWS_, 512), 256, 0, stream>>>(
      hn, wbt_v, nullptr, nullptr, vsb, MROWS_, 512);
  // 6) V^T per batch for contiguous PV B-fragments.
  transp_v<<<(BB_ * DV_ * SS_) / 256, 256, 0, stream>>>(vsb, vtb);
  // 7) Attention: 4 batches * 8 heads * 128 q-tiles.
  attn_wmma<<<BB_ * NH_ * (SS_ / 16), 256, 0, stream>>>(qsb, ksb, vtb, ao);
  // 8) h2 = f16(attn_out @ Wb_out + h)   (residual fused in epilogue).
  gemm_wmma<4096, 2><<<blocks(MROWS_, 512), 256, 0, stream>>>(
      ao, wbt_out, h, nullptr, h2, MROWS_, 512);
  // 9) out = h2 @ Wb_final (f32 -> d_out).
  gemm_wmma<512, 0><<<blocks(MROWS_, 512), 256, 0, stream>>>(
      h2, wbt_fin, nullptr, out, nullptr, MROWS_, 512);
}
